// CURDeepseekMoE_34643206210100
// MI455X (gfx1250) — compile-verified, matrix operand order checked by reading, not run on hardware
//
#include <hip/hip_runtime.h>
#include <hip/hip_bf16.h>

// ---------------------------------------------------------------------------
// CUR DeepSeek MoE forward for gfx1250 (MI455X).
// All GEMMs run on v_wmma_f32_16x16x32_bf16 (bf16 in, fp32 accumulate).
// Gate/softmax/top-k stays fp32 (discrete routing decisions).
// bf16 weights (~48MB) are L2-resident (192MB L2) -> B operands fed from
// global. Wave tile 64x32 (4x2 fragments): 8 WMMA per 12 b128 loads
// (1.5 loads/WMMA) for better FLOP per vmem-issue-slot on a compute-bound
// problem (~235 GFLOP vs ~10us of HBM traffic).
// ---------------------------------------------------------------------------

typedef __bf16 bf16_t;
typedef __attribute__((ext_vector_type(8)))  __bf16 v8bf;
typedef __attribute__((ext_vector_type(16))) __bf16 v16bf;
typedef __attribute__((ext_vector_type(8)))  float  v8f;

constexpr int Hd  = 2048;   // hidden
constexpr int Id  = 1408;   // moe intermediate
constexpr int Ed  = 32;     // experts
constexpr int SId = 2816;   // shared intermediate (I * 2)
constexpr int Td  = 4096;   // tokens (B*S)
constexpr int Rr  = 128;    // CUR rank

// ---------------------------------------------------------------------------
// WMMA helpers
// ---------------------------------------------------------------------------
__device__ __forceinline__ v8f wmma_bf16(v16bf a, v16bf b, v8f c) {
  // 8 args: (neg_a, A, neg_b, B, c_mod, C, reuse_a, reuse_b)
  return __builtin_amdgcn_wmma_f32_16x16x32_bf16(false, a, false, b,
                                                 (short)0, c, false, false);
}

// Load one 16x32 bf16 fragment (A layout; B 32x16 uses the identical pattern
// because B is stored [N,K] row-major, i.e. K-contiguous per output column).
// ISA layout: lane<16 holds row (lane&15), K = {k..k+7, k+16..k+23};
//             lane>=16 holds same row,    K = {k+8..k+15, k+24..k+31}.
// => two contiguous 16-byte loads per lane.
__device__ __forceinline__ v16bf load_frag(const bf16_t* p, int row, int ld,
                                           int k, int lane) {
  const bf16_t* q = p + (size_t)(row + (lane & 15)) * ld + k + ((lane >> 4) << 3);
  v8bf lo = *(const v8bf*)(q);
  v8bf hi = *(const v8bf*)(q + 16);
  return __builtin_shufflevector(lo, hi, 0, 1, 2, 3, 4, 5, 6, 7,
                                         8, 9, 10, 11, 12, 13, 14, 15);
}

// C/D layout: lane<16 -> rows tm+0..7, col tn+lane; lane>=16 -> rows tm+8..15.
__device__ __forceinline__ void store_frag_bf16(bf16_t* C, int ldc, int tm,
                                                int tn, v8f c, int lane) {
  int col  = tn + (lane & 15);
  int row0 = tm + ((lane >> 4) << 3);
#pragma unroll
  for (int i = 0; i < 8; ++i)
    C[(size_t)(row0 + i) * ldc + col] = (bf16_t)c[i];
}

__device__ __forceinline__ float silu_f(float x) {
  return x / (1.0f + __expf(-x));
}

// ---------------------------------------------------------------------------
// GEMM: C[M,N](bf16) = A[M,K](bf16,lda) @ B[N,K]^T(bf16,ldb)
// block = 256 thr = 8 waves (2x4), wave tile 64x32, block tile 128x128.
// grid = (M/128, N/128). Requires M%128==0, N%128==0, K%32==0.
// ---------------------------------------------------------------------------
__global__ __launch_bounds__(256) void gemm_bt_bf16_kernel(
    const bf16_t* __restrict__ A, int lda,
    const bf16_t* __restrict__ B, int ldb,
    bf16_t* __restrict__ C, int ldc, int K) {
  const int lane = threadIdx.x & 31;
  const int wave = threadIdx.x >> 5;
  const int tm = blockIdx.x * 128 + (wave >> 2) * 64;
  const int tn = blockIdx.y * 128 + (wave & 3) * 32;

  v8f c[8] = {};   // c[2*i+j]: m-group i (0..3), n-group j (0..1)
  for (int k = 0; k < K; k += 32) {
    v16bf b0 = load_frag(B, tn,      ldb, k, lane);
    v16bf b1 = load_frag(B, tn + 16, ldb, k, lane);
#pragma unroll
    for (int i = 0; i < 4; ++i) {
      v16bf a = load_frag(A, tm + 16 * i, lda, k, lane);
      c[2 * i]     = wmma_bf16(a, b0, c[2 * i]);
      c[2 * i + 1] = wmma_bf16(a, b1, c[2 * i + 1]);
    }
  }
#pragma unroll
  for (int i = 0; i < 4; ++i) {
    store_frag_bf16(C, ldc, tm + 16 * i, tn,      c[2 * i],     lane);
    store_frag_bf16(C, ldc, tm + 16 * i, tn + 16, c[2 * i + 1], lane);
  }
}

// ---------------------------------------------------------------------------
// Fused GLU: H[M,N] = silu(Ag@Bg^T) * (Au@Bu^T). K = 128 (CUR rank).
// lda = ldb = 128. Wave tile 64x32 per path. grid = (M/128, N/128).
// ---------------------------------------------------------------------------
__global__ __launch_bounds__(256) void gemm_glu_kernel(
    const bf16_t* __restrict__ Ag, const bf16_t* __restrict__ Au,
    const bf16_t* __restrict__ Bg, const bf16_t* __restrict__ Bu,
    bf16_t* __restrict__ Hout, int ldh) {
  const int lane = threadIdx.x & 31;
  const int wave = threadIdx.x >> 5;
  const int tm = blockIdx.x * 128 + (wave >> 2) * 64;
  const int tn = blockIdx.y * 128 + (wave & 3) * 32;

  v8f g[8] = {};
  v8f u[8] = {};
#pragma unroll
  for (int k = 0; k < Rr; k += 32) {
    v16bf bg0 = load_frag(Bg, tn,      Rr, k, lane);
    v16bf bg1 = load_frag(Bg, tn + 16, Rr, k, lane);
#pragma unroll
    for (int i = 0; i < 4; ++i) {
      v16bf a = load_frag(Ag, tm + 16 * i, Rr, k, lane);
      g[2 * i]     = wmma_bf16(a, bg0, g[2 * i]);
      g[2 * i + 1] = wmma_bf16(a, bg1, g[2 * i + 1]);
    }
    v16bf bu0 = load_frag(Bu, tn,      Rr, k, lane);
    v16bf bu1 = load_frag(Bu, tn + 16, Rr, k, lane);
#pragma unroll
    for (int i = 0; i < 4; ++i) {
      v16bf a = load_frag(Au, tm + 16 * i, Rr, k, lane);
      u[2 * i]     = wmma_bf16(a, bu0, u[2 * i]);
      u[2 * i + 1] = wmma_bf16(a, bu1, u[2 * i + 1]);
    }
  }
  const int col  = lane & 15;
  const int row0 = (lane >> 4) << 3;
#pragma unroll
  for (int f = 0; f < 8; ++f) {
    int rm = tm + (f >> 1) * 16 + row0;
    int cn = tn + (f & 1) * 16 + col;
#pragma unroll
    for (int i = 0; i < 8; ++i) {
      float v = silu_f(g[f][i]) * u[f][i];
      Hout[(size_t)(rm + i) * ldh + cn] = (bf16_t)v;
    }
  }
}

// ---------------------------------------------------------------------------
// Scaled accumulate: Y[M,N](f32) += w[row] * (A@B^T).
// wvec may be null (weight = 1, shared expert). wstride = E for routed.
// ---------------------------------------------------------------------------
__global__ __launch_bounds__(256) void gemm_scale_acc_kernel(
    const bf16_t* __restrict__ A, int lda,
    const bf16_t* __restrict__ B, int ldb,
    float* __restrict__ Y, int ldy, int K,
    const float* __restrict__ wvec, int wstride) {
  const int lane = threadIdx.x & 31;
  const int wave = threadIdx.x >> 5;
  const int tm = blockIdx.x * 128 + (wave >> 2) * 64;
  const int tn = blockIdx.y * 128 + (wave & 3) * 32;

  v8f c[8] = {};
  for (int k = 0; k < K; k += 32) {
    v16bf b0 = load_frag(B, tn,      ldb, k, lane);
    v16bf b1 = load_frag(B, tn + 16, ldb, k, lane);
#pragma unroll
    for (int i = 0; i < 4; ++i) {
      v16bf a = load_frag(A, tm + 16 * i, lda, k, lane);
      c[2 * i]     = wmma_bf16(a, b0, c[2 * i]);
      c[2 * i + 1] = wmma_bf16(a, b1, c[2 * i + 1]);
    }
  }
  const int col  = lane & 15;
  const int row0 = (lane >> 4) << 3;
#pragma unroll
  for (int f = 0; f < 8; ++f) {
    int rm = tm + (f >> 1) * 16 + row0;
    int cn = tn + (f & 1) * 16 + col;
#pragma unroll
    for (int i = 0; i < 8; ++i) {
      float w = wvec ? wvec[(size_t)(rm + i) * wstride] : 1.0f;
      size_t idx = (size_t)(rm + i) * ldy + cn;
      Y[idx] += w * c[f][i];
    }
  }
}

// ---------------------------------------------------------------------------
// Gate: logits (fp32) -> softmax over E=32 lanes -> top-4 -> dense we[T,E].
// One wave32 per token; lane == expert index.
// ---------------------------------------------------------------------------
__global__ __launch_bounds__(128) void gate_topk_kernel(
    const float* __restrict__ x, const float* __restrict__ gw,
    float* __restrict__ we) {
  const int lane = threadIdx.x & 31;
  const int t = blockIdx.x * 4 + (threadIdx.x >> 5);
  const float* xr = x + (size_t)t * Hd;
  const float* gr = gw + (size_t)lane * Hd;

  float acc = 0.0f;
  for (int k = 0; k < Hd; k += 4) {
    float4 xv = *(const float4*)(xr + k);
    float4 gv = *(const float4*)(gr + k);
    acc += xv.x * gv.x + xv.y * gv.y + xv.z * gv.z + xv.w * gv.w;
  }
  // softmax across the 32 lanes (= 32 experts)
  float m = acc;
  for (int o = 16; o; o >>= 1) m = fmaxf(m, __shfl_xor(m, o, 32));
  float p = __expf(acc - m);
  float s = p;
  for (int o = 16; o; o >>= 1) s += __shfl_xor(s, o, 32);
  float score = p / s;

  // top-4 by 4 rounds of ballot-argmax
  float cur = score, sel = 0.0f;
  bool kept = false;
  for (int it = 0; it < 4; ++it) {
    float mx = cur;
    for (int o = 16; o; o >>= 1) mx = fmaxf(mx, __shfl_xor(mx, o, 32));
    unsigned long long b = __ballot(cur == mx);
    int lead = __ffsll(b) - 1;
    if (lane == lead) { kept = true; sel = score; cur = -1.0f; }
  }
  float tot = kept ? sel : 0.0f;
  for (int o = 16; o; o >>= 1) tot += __shfl_xor(tot, o, 32);
  we[(size_t)t * Ed + lane] = kept ? sel / (tot + 1e-20f) : 0.0f;
}

// ---------------------------------------------------------------------------
// Elementwise utility kernels
// ---------------------------------------------------------------------------
__global__ void f32_to_bf16_kernel(const float* __restrict__ in,
                                   bf16_t* __restrict__ out, long n) {
  long i = (long)blockIdx.x * blockDim.x + threadIdx.x;
  long stride = (long)gridDim.x * blockDim.x;
  for (; i < n; i += stride) out[i] = (bf16_t)in[i];
}

__global__ void zero_f32_kernel(float* __restrict__ p, long n) {
  long i = (long)blockIdx.x * blockDim.x + threadIdx.x;
  long stride = (long)gridDim.x * blockDim.x;
  for (; i < n; i += stride) p[i] = 0.0f;
}

// ---------------------------------------------------------------------------
// Host orchestration
// ---------------------------------------------------------------------------
static inline void launch_cvt(const float* src, bf16_t* dst, long n,
                              hipStream_t stream) {
  long blocks = (n + 255) / 256;
  if (blocks > 8192) blocks = 8192;
  f32_to_bf16_kernel<<<(int)blocks, 256, 0, stream>>>(src, dst, n);
}

extern "C" void kernel_launch(void* const* d_in, const int* in_sizes, int n_in,
                              void* d_out, int out_size, void* d_ws,
                              size_t ws_size, hipStream_t stream) {
  (void)in_sizes; (void)n_in; (void)out_size; (void)ws_size;

  const float* x    = (const float*)d_in[0];
  const float* gw   = (const float*)d_in[1];
  const float* Rg   = (const float*)d_in[2];
  const float* Ru   = (const float*)d_in[3];
  const float* Rd   = (const float*)d_in[4];
  const float* Ug   = (const float*)d_in[5];
  const float* Cg   = (const float*)d_in[6];
  const float* Uu   = (const float*)d_in[7];
  const float* Cu   = (const float*)d_in[8];
  const float* Ud   = (const float*)d_in[9];
  const float* Cd   = (const float*)d_in[10];
  const float* sRg  = (const float*)d_in[11];
  const float* sUg  = (const float*)d_in[12];
  const float* sCg  = (const float*)d_in[13];
  const float* sRu  = (const float*)d_in[14];
  const float* sUu  = (const float*)d_in[15];
  const float* sCu  = (const float*)d_in[16];
  const float* sRd  = (const float*)d_in[17];
  const float* sUd  = (const float*)d_in[18];
  const float* sCd  = (const float*)d_in[19];
  float* y = (float*)d_out;

  // ---- workspace carve (bf16 mirrors + activations), 256B aligned ----
  char* base = (char*)d_ws;
  size_t off = 0;
  auto alloc = [&](size_t bytes) -> char* {
    char* p = base + off;
    off += (bytes + 255) & ~(size_t)255;
    return p;
  };
  bf16_t* xb   = (bf16_t*)alloc((size_t)Td * Hd * 2);
  bf16_t* Rb   = (bf16_t*)alloc((size_t)2 * Rr * Hd * 2);   // [Rg;Ru] 256xH
  bf16_t* Rdb  = (bf16_t*)alloc((size_t)Rr * Id * 2);
  bf16_t* Ugb  = (bf16_t*)alloc((size_t)Ed * Rr * Rr * 2);
  bf16_t* Uub  = (bf16_t*)alloc((size_t)Ed * Rr * Rr * 2);
  bf16_t* Udb  = (bf16_t*)alloc((size_t)Ed * Rr * Rr * 2);
  bf16_t* Cgb  = (bf16_t*)alloc((size_t)Ed * Id * Rr * 2);
  bf16_t* Cub  = (bf16_t*)alloc((size_t)Ed * Id * Rr * 2);
  bf16_t* Cdb  = (bf16_t*)alloc((size_t)Ed * Hd * Rr * 2);
  bf16_t* sRb  = (bf16_t*)alloc((size_t)2 * Rr * Hd * 2);   // [s_Rg;s_Ru]
  bf16_t* sUgb = (bf16_t*)alloc((size_t)Rr * Rr * 2);
  bf16_t* sUub = (bf16_t*)alloc((size_t)Rr * Rr * 2);
  bf16_t* sUdb = (bf16_t*)alloc((size_t)Rr * Rr * 2);
  bf16_t* sCgb = (bf16_t*)alloc((size_t)SId * Rr * 2);
  bf16_t* sCub = (bf16_t*)alloc((size_t)SId * Rr * 2);
  bf16_t* sRdb = (bf16_t*)alloc((size_t)Rr * SId * 2);
  bf16_t* sCdb = (bf16_t*)alloc((size_t)Hd * Rr * 2);
  float*  we   = (float*)alloc((size_t)Td * Ed * 4);
  bf16_t* rgru = (bf16_t*)alloc((size_t)Td * 2 * Rr * 2);   // [T,256]
  bf16_t* srgru= (bf16_t*)alloc((size_t)Td * 2 * Rr * 2);
  bf16_t* tg   = (bf16_t*)alloc((size_t)Td * Rr * 2);
  bf16_t* tu   = (bf16_t*)alloc((size_t)Td * Rr * 2);
  bf16_t* hbuf = (bf16_t*)alloc((size_t)Td * SId * 2);
  bf16_t* rdb  = (bf16_t*)alloc((size_t)Td * Rr * 2);
  bf16_t* tdb  = (bf16_t*)alloc((size_t)Td * Rr * 2);

  // ---- init output, convert operands to bf16 ----
  zero_f32_kernel<<<4096, 256, 0, stream>>>(y, (long)Td * Hd);
  launch_cvt(x,   xb,   (long)Td * Hd, stream);
  launch_cvt(Rg,  Rb,               (long)Rr * Hd, stream);
  launch_cvt(Ru,  Rb + (size_t)Rr * Hd, (long)Rr * Hd, stream);
  launch_cvt(Rd,  Rdb,  (long)Rr * Id, stream);
  launch_cvt(Ug,  Ugb,  (long)Ed * Rr * Rr, stream);
  launch_cvt(Uu,  Uub,  (long)Ed * Rr * Rr, stream);
  launch_cvt(Ud,  Udb,  (long)Ed * Rr * Rr, stream);
  launch_cvt(Cg,  Cgb,  (long)Ed * Id * Rr, stream);
  launch_cvt(Cu,  Cub,  (long)Ed * Id * Rr, stream);
  launch_cvt(Cd,  Cdb,  (long)Ed * Hd * Rr, stream);
  launch_cvt(sRg, sRb,              (long)Rr * Hd, stream);
  launch_cvt(sRu, sRb + (size_t)Rr * Hd, (long)Rr * Hd, stream);
  launch_cvt(sUg, sUgb, (long)Rr * Rr, stream);
  launch_cvt(sUu, sUub, (long)Rr * Rr, stream);
  launch_cvt(sUd, sUdb, (long)Rr * Rr, stream);
  launch_cvt(sCg, sCgb, (long)SId * Rr, stream);
  launch_cvt(sCu, sCub, (long)SId * Rr, stream);
  launch_cvt(sRd, sRdb, (long)Rr * SId, stream);
  launch_cvt(sCd, sCdb, (long)Hd * Rr, stream);

  // ---- gate: fp32 softmax + top-4 -> dense combine weights we[T,E] ----
  gate_topk_kernel<<<Td / 4, 128, 0, stream>>>(x, gw, we);

  const dim3 blk(256);
  const dim3 gM(Td / 128, 1);

  // ---- rank projections: [rg|ru] = x @ [Rg;Ru]^T  (T x 256) ----
  gemm_bt_bf16_kernel<<<dim3(Td / 128, 256 / 128), blk, 0, stream>>>(
      xb, Hd, Rb, Hd, rgru, 2 * Rr, Hd);
  gemm_bt_bf16_kernel<<<dim3(Td / 128, 256 / 128), blk, 0, stream>>>(
      xb, Hd, sRb, Hd, srgru, 2 * Rr, Hd);

  // ---- routed experts ----
  for (int e = 0; e < Ed; ++e) {
    const bf16_t* Ug_e = Ugb + (size_t)e * Rr * Rr;
    const bf16_t* Uu_e = Uub + (size_t)e * Rr * Rr;
    const bf16_t* Ud_e = Udb + (size_t)e * Rr * Rr;
    const bf16_t* Cg_e = Cgb + (size_t)e * Id * Rr;
    const bf16_t* Cu_e = Cub + (size_t)e * Id * Rr;
    const bf16_t* Cd_e = Cdb + (size_t)e * Hd * Rr;

    // tg = rg @ Ug^T ; tu = ru @ Uu^T        (T x 128, K=128)
    gemm_bt_bf16_kernel<<<gM, blk, 0, stream>>>(rgru, 2 * Rr, Ug_e, Rr,
                                                tg, Rr, Rr);
    gemm_bt_bf16_kernel<<<gM, blk, 0, stream>>>(rgru + Rr, 2 * Rr, Uu_e, Rr,
                                                tu, Rr, Rr);
    // h = silu(tg @ Cg^T) * (tu @ Cu^T)      (T x I)
    gemm_glu_kernel<<<dim3(Td / 128, Id / 128), blk, 0, stream>>>(
        tg, tu, Cg_e, Cu_e, hbuf, Id);
    // rd = h @ Rd^T                          (T x 128, K=I)
    gemm_bt_bf16_kernel<<<gM, blk, 0, stream>>>(hbuf, Id, Rdb, Id,
                                                rdb, Rr, Id);
    // td = rd @ Ud^T                         (T x 128, K=128)
    gemm_bt_bf16_kernel<<<gM, blk, 0, stream>>>(rdb, Rr, Ud_e, Rr,
                                                tdb, Rr, Rr);
    // y += we[:,e] * (td @ Cd^T)             (T x H)
    gemm_scale_acc_kernel<<<dim3(Td / 128, Hd / 128), blk, 0, stream>>>(
        tdb, Rr, Cd_e, Rr, y, Hd, Rr, we + e, Ed);
  }

  // ---- shared experts (weight = 1, intermediate = SI) ----
  gemm_bt_bf16_kernel<<<gM, blk, 0, stream>>>(srgru, 2 * Rr, sUgb, Rr,
                                              tg, Rr, Rr);
  gemm_bt_bf16_kernel<<<gM, blk, 0, stream>>>(srgru + Rr, 2 * Rr, sUub, Rr,
                                              tu, Rr, Rr);
  gemm_glu_kernel<<<dim3(Td / 128, SId / 128), blk, 0, stream>>>(
      tg, tu, sCgb, sCub, hbuf, SId);
  gemm_bt_bf16_kernel<<<gM, blk, 0, stream>>>(hbuf, SId, sRdb, SId,
                                              rdb, Rr, SId);
  gemm_bt_bf16_kernel<<<gM, blk, 0, stream>>>(rdb, Rr, sUdb, Rr,
                                              tdb, Rr, Rr);
  gemm_scale_acc_kernel<<<dim3(Td / 128, Hd / 128), blk, 0, stream>>>(
      tdb, Rr, sCdb, Rr, y, Hd, Rr, nullptr, 0);
}
